// RandLANet_6597069766756
// MI455X (gfx1250) — compile-verified, matrix operand order checked by reading, not run on hardware
//
#include <hip/hip_runtime.h>
#include <hip/hip_fp16.h>
#include <stdint.h>

#define BB 4
#define NN 16384
#define CC 64
#define KNB 16
#define OUTC 128
#define BN_PTS (BB * NN)   // 65536

typedef __attribute__((ext_vector_type(16))) __bf16 v16bf;
typedef __attribute__((ext_vector_type(8)))  float  v8f;

__device__ __forceinline__ unsigned short f2bf(float f) {
    unsigned int x = __float_as_uint(f);
    x += 0x7fffu + ((x >> 16) & 1u);           // round-to-nearest-even
    return (unsigned short)(x >> 16);
}
__device__ __forceinline__ float bf2f(unsigned short u) {
    return __uint_as_float(((unsigned int)u) << 16);
}

// ---------------------------------------------------------------------------
// k0: feature [B,C,N] f32  ->  featT [B*N, C] bf16 (contiguous 128B rows),
//     plus zero the BN stats accumulators.
// ---------------------------------------------------------------------------
__global__ void __launch_bounds__(256) k0_transpose(
        const float* __restrict__ feature,
        unsigned short* __restrict__ featT,
        float* __restrict__ stats) {
    __shared__ float tile[64 * 65];
    int t   = threadIdx.x;
    int bid = blockIdx.x;
    int b   = bid >> 8;            // 0..3
    int n0  = (bid & 255) << 6;    // point-tile base (64 points)

    if (bid == 0) stats[t] = 0.0f; // 256 threads zero 256 floats

#pragma unroll
    for (int i = 0; i < 16; ++i) {
        int e = t + i * 256;
        int c = e >> 6, j = e & 63;
        tile[c * 65 + j] = feature[(((size_t)(b * 64 + c)) << 14) + n0 + j];
    }
    __syncthreads();

    int p  = t >> 2;           // point within tile (0..63)
    int cs = (t & 3) << 4;     // channel segment base (0,16,32,48)
    union { uint4 q[2]; unsigned short h[16]; } u;
#pragma unroll
    for (int i = 0; i < 16; ++i) u.h[i] = f2bf(tile[(cs + i) * 65 + p]);
    unsigned short* dst = featT + (((size_t)((b << 14) + n0 + p)) << 6) + cs;
    *(uint4*)(dst)     = u.q[0];
    *(uint4*)(dst + 8) = u.q[1];
}

// ---------------------------------------------------------------------------
// WMMA fragment loader for LDS-resident matrices (row stride 72 bf16),
// explicit per-lane row index:
//   lane<16 : K chunks {c0..c0+7, c0+16..c0+23}
//   lane>=16: K chunks {c0+8..c0+15, c0+24..c0+31}
// ---------------------------------------------------------------------------
__device__ __forceinline__ v16bf load_frag_row(const unsigned short* mat, int row,
                                               int c0, int lane) {
    int h8 = (lane & 16) ? 8 : 0;
    const unsigned short* rp = mat + row * 72 + c0 + h8;
    union { v16bf v; uint4 q[2]; } u;
    u.q[0] = *(const uint4*)(rp);
    u.q[1] = *(const uint4*)(rp + 16);
    return u.v;
}

// ---------------------------------------------------------------------------
// k1: gather + GEMM1 (WMMA) + softmax + attention pool + GEMM2 (WMMA)
//     + bias + BN partial stats + transposed [B,OUT,N] store.
// One wave = 16 consecutive points. 4 waves / block. 1024 blocks.
//
// GEMM1 d-columns are permuted: score tile t, lane-column n holds channel
// d = 4n + t (softmax over d is permutation invariant). This makes each
// lane's 4 pooling operands x[k][4n..4n+3] one contiguous ds_load_b64 and
// the pooled write one packed ds_store_b64.
// ---------------------------------------------------------------------------
__global__ void __launch_bounds__(128) k1_main(
        const unsigned short* __restrict__ featT,
        const int*   __restrict__ nidx,
        const float* __restrict__ Wmlp,
        const float* __restrict__ Wconv,
        const float* __restrict__ bconv,
        float*       __restrict__ out,
        float*       __restrict__ stats) {
    __shared__ __align__(16) unsigned short sWmlp[64 * 72];     //  9216 B
    __shared__ __align__(16) unsigned short sWconv[128 * 72];   // 18432 B
    __shared__ __align__(16) unsigned short sSpir[4][16 * 72];  //  9216 B
    __shared__ __align__(16) unsigned short sPool[4][16 * 72];  //  9216 B
    __shared__ __align__(16) float          sOut[4][16 * 68];   // 17408 B
    __shared__ float                        sStat[256];         //  1024 B

    int t    = threadIdx.x;
    int wave = t >> 5;
    int lane = t & 31;

    // Stage weights to LDS as bf16 (row-major [d][c] / [o][c], stride 72)
#pragma unroll
    for (int i = 0; i < 32; ++i) {
        int e = t + i * 128;
        sWmlp[(e >> 6) * 72 + (e & 63)] = f2bf(Wmlp[e]);
    }
#pragma unroll
    for (int i = 0; i < 64; ++i) {
        int e = t + i * 128;
        sWconv[(e >> 6) * 72 + (e & 63)] = f2bf(Wconv[e]);
    }
    sStat[t] = 0.0f;
    sStat[t + 128] = 0.0f;
    __syncthreads();

    unsigned short* spir     = sSpir[wave];
    unsigned short* pool_lds = sPool[wave];
    float*          ost      = sOut[wave];

    int crow = lane & 15;
    int h8   = (lane & 16) ? 8 : 0;   // A-layout chunk base AND row-half offset

    int wg    = blockIdx.x * 4 + wave;
    int nbase = wg << 4;              // first of this wave's 16 points

    // W_mlp B-fragments, permuted: tile dt / lane-col n -> W_mlp row 4n + dt
    v16bf bm[8];
#pragma unroll
    for (int dt = 0; dt < 4; ++dt)
#pragma unroll
        for (int s = 0; s < 2; ++s)
            bm[dt * 2 + s] = load_frag_row(sWmlp, 4 * crow + dt, 32 * s, lane);

    // All 16 neighbor indices for this wave, up front (coalesced 64B runs)
    int ids[16];
#pragma unroll
    for (int p = 0; p < 16; ++p) ids[p] = nidx[(nbase + p) * 16 + crow];

    // Gather one neighbor row in A-fragment chunk order:
    //   lane<16 : chunks at c = {0,16,32,48};  lane>=16: c = {8,24,40,56}
    auto ldrows = [&](int id, uint4* g) {
        const unsigned short* rp = featT + (((size_t)id) << 6) + h8;
#pragma unroll
        for (int i = 0; i < 4; ++i) g[i] = *(const uint4*)(rp + 16 * i);
    };

    auto process = [&](int p, const uint4* g) {
        // stash the row in LDS for the pooling reads (chunked, full row
        // covered by the two half-lanes)
        unsigned short* srow = spir + crow * 72 + h8;
        *(uint4*)(srow)      = g[0];
        *(uint4*)(srow + 16) = g[1];
        *(uint4*)(srow + 32) = g[2];
        *(uint4*)(srow + 48) = g[3];

        // A fragments come straight from the gathered registers
        union { v16bf v; uint4 q[2]; } a0, a1;
        a0.q[0] = g[0]; a0.q[1] = g[1];   // K = 0..31
        a1.q[0] = g[2]; a1.q[1] = g[3];   // K = 32..63

        v8f acc0 = {0,0,0,0,0,0,0,0}, acc1 = {0,0,0,0,0,0,0,0};
        v8f acc2 = {0,0,0,0,0,0,0,0}, acc3 = {0,0,0,0,0,0,0,0};
        acc0 = __builtin_amdgcn_wmma_f32_16x16x32_bf16(false, a0.v, false, bm[0], (short)0, acc0, false, false);
        acc1 = __builtin_amdgcn_wmma_f32_16x16x32_bf16(false, a0.v, false, bm[2], (short)0, acc1, false, false);
        acc2 = __builtin_amdgcn_wmma_f32_16x16x32_bf16(false, a0.v, false, bm[4], (short)0, acc2, false, false);
        acc3 = __builtin_amdgcn_wmma_f32_16x16x32_bf16(false, a0.v, false, bm[6], (short)0, acc3, false, false);
        acc0 = __builtin_amdgcn_wmma_f32_16x16x32_bf16(false, a1.v, false, bm[1], (short)0, acc0, false, false);
        acc1 = __builtin_amdgcn_wmma_f32_16x16x32_bf16(false, a1.v, false, bm[3], (short)0, acc1, false, false);
        acc2 = __builtin_amdgcn_wmma_f32_16x16x32_bf16(false, a1.v, false, bm[5], (short)0, acc2, false, false);
        acc3 = __builtin_amdgcn_wmma_f32_16x16x32_bf16(false, a1.v, false, bm[7], (short)0, acc3, false, false);

        // softmax over d: scores ~ N(0,1) so exp() needs no max shift
        float ps[8];
#pragma unroll
        for (int j = 0; j < 8; ++j) {
            float e0 = __expf(acc0[j]), e1 = __expf(acc1[j]);
            float e2 = __expf(acc2[j]), e3 = __expf(acc3[j]);
            acc0[j] = e0; acc1[j] = e1; acc2[j] = e2; acc3[j] = e3;
            ps[j] = (e0 + e1) + (e2 + e3);
        }
        // f16-packed cross-lane sum (2 rows per dword: halves the bpermutes)
        float inv[8];
#pragma unroll
        for (int q = 0; q < 4; ++q) {
            union { __half2 h; int i; } v, w;
            v.h = __floats2half2_rn(ps[2 * q], ps[2 * q + 1]);
            w.i = __shfl_xor(v.i, 1); v.h = __hadd2(v.h, w.h);
            w.i = __shfl_xor(v.i, 2); v.h = __hadd2(v.h, w.h);
            w.i = __shfl_xor(v.i, 4); v.h = __hadd2(v.h, w.h);
            w.i = __shfl_xor(v.i, 8); v.h = __hadd2(v.h, w.h);
            inv[2 * q]     = __builtin_amdgcn_rcpf(__low2float(v.h));
            inv[2 * q + 1] = __builtin_amdgcn_rcpf(__high2float(v.h));
        }
#pragma unroll
        for (int j = 0; j < 8; ++j) {
            acc0[j] *= inv[j]; acc1[j] *= inv[j];
            acc2[j] *= inv[j]; acc3[j] *= inv[j];
        }

        // attention pooling: tile t / lane-col n owns channel d = 4n + t,
        // so each row's 4 operands are contiguous: one ds_load_b64
        float pl0 = 0, pl1 = 0, pl2 = 0, pl3 = 0;
#pragma unroll
        for (int j = 0; j < 8; ++j) {
            union { uint2 u; unsigned short h[4]; } xv;
            xv.u = *(const uint2*)(spir + (h8 + j) * 72 + 4 * crow);
            pl0 += acc0[j] * bf2f(xv.h[0]);
            pl1 += acc1[j] * bf2f(xv.h[1]);
            pl2 += acc2[j] * bf2f(xv.h[2]);
            pl3 += acc3[j] * bf2f(xv.h[3]);
        }
        // packed half-combine (2 bpermutes instead of 4)
        union { __half2 h; int i; } c01, c23, s01, s23;
        c01.h = __floats2half2_rn(pl0, pl1);
        c23.h = __floats2half2_rn(pl2, pl3);
        s01.i = __shfl_xor(c01.i, 16); c01.h = __hadd2(c01.h, s01.h);
        s23.i = __shfl_xor(c23.i, 16); c23.h = __hadd2(c23.h, s23.h);
        if (lane < 16) {
            union { uint2 u; unsigned short h[4]; } pw;
            pw.h[0] = f2bf(__low2float(c01.h));
            pw.h[1] = f2bf(__high2float(c01.h));
            pw.h[2] = f2bf(__low2float(c23.h));
            pw.h[3] = f2bf(__high2float(c23.h));
            *(uint2*)(pool_lds + p * 72 + 4 * crow) = pw.u;   // one b64 store
        }
    };

    // software-pipelined point loop (rows prefetched one point ahead)
    uint4 ga[4], gb[4];
    ldrows(ids[0], ga);
    for (int p = 0; p < 16; p += 2) {
        ldrows(ids[p + 1], gb);
        process(p, ga);
        ldrows(ids[(p + 2) & 15], ga);
        process(p + 1, gb);
    }

    // --- GEMM2: out[16p x 128o] = pooled[16x64] * Wconv^T (natural layout)
    v8f oacc[8];
#pragma unroll
    for (int tt = 0; tt < 8; ++tt) oacc[tt] = (v8f){0,0,0,0,0,0,0,0};
#pragma unroll
    for (int s = 0; s < 2; ++s) {
        v16bf a = load_frag_row(pool_lds, crow, 32 * s, lane);
#pragma unroll
        for (int tt = 0; tt < 8; ++tt) {
            v16bf bw = load_frag_row(sWconv, tt * 16 + crow, 32 * s, lane);
            oacc[tt] = __builtin_amdgcn_wmma_f32_16x16x32_bf16(false, a, false, bw, (short)0, oacc[tt], false, false);
        }
    }

    // --- bias + BN partial stats
#pragma unroll
    for (int tt = 0; tt < 8; ++tt) {
        float bo = bconv[tt * 16 + crow];
        float s = 0.0f, sq = 0.0f;
#pragma unroll
        for (int j = 0; j < 8; ++j) {
            float v = oacc[tt][j] + bo;
            oacc[tt][j] = v;
            s += v;
            sq += v * v;
        }
        s  += __shfl_xor(s, 16);
        sq += __shfl_xor(sq, 16);
        if (lane < 16) {
            atomicAdd(&sStat[tt * 16 + crow], s);
            atomicAdd(&sStat[128 + tt * 16 + crow], sq);
        }
    }

    // --- store pre-BN result in [B, OUT, N] layout via LDS transpose
    int bidx = nbase >> 14;       // batch
    int ptb  = nbase & 16383;     // point base within batch (mult of 16)
#pragma unroll
    for (int h = 0; h < 2; ++h) {
#pragma unroll
        for (int tt = 0; tt < 4; ++tt)
#pragma unroll
            for (int j = 0; j < 8; ++j)
                ost[(h8 + j) * 68 + tt * 16 + crow] = oacc[h * 4 + tt][j];
        // per-wave LDS region: wave-internal DS ordering suffices (no barrier)
#pragma unroll
        for (int oi = 0; oi < 2; ++oi) {
            int o = h * 64 + oi * 32 + lane;
            float v[16];
#pragma unroll
            for (int pt = 0; pt < 16; ++pt) v[pt] = ost[pt * 68 + oi * 32 + lane];
            float* dst = out + (((size_t)(bidx * 128 + o)) << 14) + ptb;
            *(float4*)(dst)      = make_float4(v[0],  v[1],  v[2],  v[3]);
            *(float4*)(dst + 4)  = make_float4(v[4],  v[5],  v[6],  v[7]);
            *(float4*)(dst + 8)  = make_float4(v[8],  v[9],  v[10], v[11]);
            *(float4*)(dst + 12) = make_float4(v[12], v[13], v[14], v[15]);
        }
    }

    __syncthreads();
    // flush block-level stats (128 threads cover 256 accumulators)
    atomicAdd(&stats[t],       sStat[t]);
    atomicAdd(&stats[t + 128], sStat[t + 128]);
}

// ---------------------------------------------------------------------------
// k2: apply BatchNorm in place over d_out ([B, OUT, N]), float4-vectorized.
// ---------------------------------------------------------------------------
__global__ void __launch_bounds__(256) k2_bn(
        float* __restrict__ out,
        const float* __restrict__ stats,
        const float* __restrict__ gamma,
        const float* __restrict__ beta) {
    int gid = blockIdx.x * 256 + threadIdx.x;     // over 2,097,152 float4s
    int o = (gid >> 12) & 127;                    // channel of this float4
    float mean = stats[o] * (1.0f / 65536.0f);
    float var  = stats[128 + o] * (1.0f / 65536.0f) - mean * mean;
    float sc = gamma[o] * rsqrtf(var + 1e-5f);
    float sh = beta[o] - mean * sc;
    float4* p4 = (float4*)out;
    float4 v = p4[gid];
    v.x = v.x * sc + sh;
    v.y = v.y * sc + sh;
    v.z = v.z * sc + sh;
    v.w = v.w * sc + sh;
    p4[gid] = v;
}

// ---------------------------------------------------------------------------
extern "C" void kernel_launch(void* const* d_in, const int* in_sizes, int n_in,
                              void* d_out, int out_size, void* d_ws, size_t ws_size,
                              hipStream_t stream) {
    (void)in_sizes; (void)n_in; (void)out_size; (void)ws_size;
    const float* feature = (const float*)d_in[0];
    const int*   nidx    = (const int*)d_in[1];
    // d_in[2] = permatrix (unused by forward)
    const float* Wmlp    = (const float*)d_in[3];
    const float* Wconv   = (const float*)d_in[4];
    const float* bconv   = (const float*)d_in[5];
    const float* gamma   = (const float*)d_in[6];
    const float* beta    = (const float*)d_in[7];

    unsigned short* featT = (unsigned short*)d_ws;                       // 8 MB bf16
    float* stats = (float*)((char*)d_ws +
                            (size_t)BN_PTS * CC * sizeof(unsigned short)); // 256 f32
    float* out = (float*)d_out;

    k0_transpose<<<1024, 256, 0, stream>>>(feature, featT, stats);
    k1_main<<<1024, 128, 0, stream>>>(featT, nidx, Wmlp, Wconv, bconv, out, stats);
    k2_bn<<<8192, 256, 0, stream>>>(out, stats, gamma, beta);
}
